// SelfAttnPool_48842368090703
// MI455X (gfx1250) — matrix operand order repaired and checked by reading, not compiled, not run on hardware
//
#include <hip/hip_runtime.h>
#include <hip/hip_bf16.h>

typedef __attribute__((ext_vector_type(16))) __bf16 v16bf;
typedef __attribute__((ext_vector_type(8)))  float  v8f;
typedef __attribute__((ext_vector_type(4)))  int    v4i;

#define B_  32
#define T_  4096
#define F_  512
#define U_  512
#define ROWP 20                    // LDS row stride in bf16 elems (40B -> conflict-friendly)
#define SLAB_HALVES (F_ * ROWP)    // one N-tile slab: 512 rows * 20 halves
#define SLAB_BYTES  (SLAB_HALVES * 2)   // 20480 B

#if defined(__gfx1250__) && __has_builtin(__builtin_amdgcn_global_load_async_to_lds_b128) && __has_builtin(__builtin_amdgcn_s_wait_asynccnt)
#define SAP_ASYNC_LDS 1
#else
#define SAP_ASYNC_LDS 0
#endif

static __device__ __forceinline__ __bf16 f2bf(float f) {
  // round-to-nearest-even fp32 -> bf16
  unsigned u = __builtin_bit_cast(unsigned, f);
  unsigned r = u + 0x7FFFu + ((u >> 16) & 1u);
  unsigned short h = (unsigned short)(r >> 16);
  return __builtin_bit_cast(__bf16, h);
}

static __device__ __forceinline__ float fast_tanh(float x) {
#if defined(__gfx1250__) && __has_builtin(__builtin_amdgcn_tanhf)
  return __builtin_amdgcn_tanhf(x);      // single V_TANH_F32 transcendental
#else
  return tanhf(x);
#endif
}

// ---------------------------------------------------------------------------
// Kernel 0: one-shot convert W (F x U fp32) into bf16 slabs laid out exactly
// as the LDS B-tile: wbf[nt][k*ROWP + n] = bf16(W[k][nt*16+n]).
// ---------------------------------------------------------------------------
__global__ __launch_bounds__(256)
void sap_wconv_kernel(const float* __restrict__ W, __bf16* __restrict__ wbf) {
  const int idx = blockIdx.x * 256 + threadIdx.x;    // over F_*U_
  if (idx >= F_ * U_) return;
  const int k  = idx >> 9;        // / U_
  const int u  = idx & (U_ - 1);
  const int nt = u >> 4, n = u & 15;
  wbf[(size_t)nt * SLAB_HALVES + k * ROWP + n] = f2bf(W[idx]);
}

// ---------------------------------------------------------------------------
// Kernel 1: scores[m] = tanh(x[m,:] @ W + bias) @ V   for m in [0, B*T)
// 8 waves/block, each wave computes a 16-row strip with v_wmma_f32_16x16x32_bf16
// ---------------------------------------------------------------------------
__global__ __launch_bounds__(256)
void sap_scores_kernel(const float* __restrict__ x, const __bf16* __restrict__ wbf,
                       const float* __restrict__ bias, const float* __restrict__ V,
                       float* __restrict__ scores) {
  __shared__ __align__(16) __bf16 wslice[SLAB_HALVES];   // 20480 bytes

  const int tid  = threadIdx.x;
  const int wave = tid >> 5;
  const int lane = tid & 31;
  const int m0   = blockIdx.x * 128 + wave * 16;
  const int mrow = lane & 15;    // A: M index / D: N index
  const int hi   = lane >> 4;

  // --- Preload A fragments: 16 rows x K=512, bf16, ISA 16-bit A layout.
  // lane (0-15): halves 0..7 -> K=k0+0..7,   halves 8..15 -> K=k0+16..23
  // lane (16-31): same +8
  v16bf afrag[16];
  {
    const float* xrow = x + (size_t)(m0 + mrow) * F_;
    #pragma unroll
    for (int kt = 0; kt < 16; ++kt) {
      const int kbase = kt * 32 + hi * 8;
      #pragma unroll
      for (int j = 0; j < 8; ++j) {
        afrag[kt][j]     = f2bf(xrow[kbase + j]);
        afrag[kt][j + 8] = f2bf(xrow[kbase + 16 + j]);
      }
    }
  }

  float srow[8];
  #pragma unroll
  for (int r = 0; r < 8; ++r) srow[r] = 0.f;

  for (int nt = 0; nt < U_ / 16; ++nt) {
    const int n0 = nt * 16;

    // --- Stage pre-converted bf16 W tile (contiguous 20 KB) into LDS.
    __syncthreads();   // previous slice fully consumed
#if SAP_ASYNC_LDS
    {
      const char* gbase = (const char*)wbf + (size_t)nt * SLAB_BYTES;
      for (int i = tid; i < SLAB_BYTES / 16; i += 256) {
        v4i* gsrc = (v4i*)(gbase + i * 16);                 // global src
        v4i* ldst = (v4i*)((char*)wslice + i * 16);         // LDS dst
        __builtin_amdgcn_global_load_async_to_lds_b128(gsrc, ldst, 0, 0);
      }
      __builtin_amdgcn_s_wait_asynccnt(0);
    }
#else
    {
      const uint4* src4 = (const uint4*)((const char*)wbf + (size_t)nt * SLAB_BYTES);
      uint4* dst4 = (uint4*)wslice;
      for (int i = tid; i < SLAB_BYTES / 16; i += 256)
        dst4[i] = src4[i];
    }
#endif
    __syncthreads();

    // --- K loop: 16 chained WMMAs, C/D fp32 accumulator.
    v8f c = {};
    #pragma unroll
    for (int kt = 0; kt < 16; ++kt) {
      // B layout: lane%16 = N, halves 0..7 -> K=8*hi+0..7, halves 8..15 -> K=16+8*hi+0..7
      const __bf16* bp = &wslice[(kt * 32 + hi * 8) * ROWP + mrow];
      v16bf bfrag;
      #pragma unroll
      for (int j = 0; j < 16; ++j) bfrag[j] = bp[(j + (j >> 3) * 8) * ROWP];
      c = __builtin_amdgcn_wmma_f32_16x16x32_bf16(
              /*neg_a=*/false, afrag[kt], /*neg_b=*/false, bfrag,
              /*c_mod=*/(short)0, c, /*reuse_a=*/false, /*reuse_b=*/false);
    }

    // --- epilogue: tanh(+bias) * V[n], accumulate per-row partials.
    // D layout: c[r] = tile[M = r + 8*hi][N = mrow]
    const float bn = bias[n0 + mrow];
    const float vn = V[n0 + mrow];
    #pragma unroll
    for (int r = 0; r < 8; ++r)
      srow[r] += fast_tanh(c[r] + bn) * vn;
  }

  // Reduce over the 16 lanes sharing each row (N dimension), wave32 shuffles.
  #pragma unroll
  for (int r = 0; r < 8; ++r) {
    float v = srow[r];
    v += __shfl_xor(v, 1, 16);
    v += __shfl_xor(v, 2, 16);
    v += __shfl_xor(v, 4, 16);
    v += __shfl_xor(v, 8, 16);
    srow[r] = v;
  }
  if (mrow == 0) {                      // lane 0 -> rows 0..7, lane 16 -> rows 8..15
    #pragma unroll
    for (int r = 0; r < 8; ++r)
      scores[m0 + hi * 8 + r] = srow[r];
  }
}

// ---------------------------------------------------------------------------
// Kernel 2: softmax over T per batch
// ---------------------------------------------------------------------------
__global__ __launch_bounds__(256)
void sap_softmax_kernel(const float* __restrict__ scores, float* __restrict__ attn) {
  __shared__ float red[256];
  const int b   = blockIdx.x;
  const int tid = threadIdx.x;
  const float* s = scores + (size_t)b * T_;
  float*       a = attn   + (size_t)b * T_;

  float m = -3.402823466e+38f;
  for (int t = tid; t < T_; t += 256) m = fmaxf(m, s[t]);
  red[tid] = m; __syncthreads();
  for (int off = 128; off > 0; off >>= 1) {
    if (tid < off) red[tid] = fmaxf(red[tid], red[tid + off]);
    __syncthreads();
  }
  m = red[0]; __syncthreads();

  float sum = 0.f;
  for (int t = tid; t < T_; t += 256) {
    float e = __expf(s[t] - m);
    a[t] = e;
    sum += e;
  }
  red[tid] = sum; __syncthreads();
  for (int off = 128; off > 0; off >>= 1) {
    if (tid < off) red[tid] += red[tid + off];
    __syncthreads();
  }
  const float inv = 1.f / red[0];
  for (int t = tid; t < T_; t += 256) a[t] *= inv;
}

// ---------------------------------------------------------------------------
// Kernel 3: out[b,f] = sum_t x[b,t,f] * attn[b,t]   (float4 streaming)
// ---------------------------------------------------------------------------
__global__ __launch_bounds__(128)
void sap_pool_kernel(const float* __restrict__ x, const float* __restrict__ attn,
                     float* __restrict__ out) {
  __shared__ float aLds[T_];            // 16 KB
  const int b   = blockIdx.x;
  const int tid = threadIdx.x;
  for (int t = tid; t < T_; t += 128) aLds[t] = attn[(size_t)b * T_ + t];
  __syncthreads();

  const float4* xp = (const float4*)(x + (size_t)b * T_ * F_);
  float4 acc = make_float4(0.f, 0.f, 0.f, 0.f);
  for (int t = 0; t < T_; ++t) {
    float4 v = xp[(size_t)t * (F_ / 4) + tid];
    float  w = aLds[t];
    acc.x += v.x * w; acc.y += v.y * w; acc.z += v.z * w; acc.w += v.w * w;
  }
  ((float4*)out)[b * (F_ / 4) + tid] = acc;
}

// ---------------------------------------------------------------------------
extern "C" void kernel_launch(void* const* d_in, const int* in_sizes, int n_in,
                              void* d_out, int out_size, void* d_ws, size_t ws_size,
                              hipStream_t stream) {
  (void)in_sizes; (void)n_in; (void)out_size; (void)ws_size;
  const float* x    = (const float*)d_in[0];
  const float* W    = (const float*)d_in[1];
  const float* bias = (const float*)d_in[2];
  const float* V    = (const float*)d_in[3];
  float* out = (float*)d_out;

  float*  scores = (float*)d_ws;                     // B*T floats (512 KB)
  float*  attn   = scores + (size_t)B_ * T_;         // B*T floats (512 KB)
  __bf16* wbf    = (__bf16*)(attn + (size_t)B_ * T_); // 32 slabs * 20480 B = 640 KB

  sap_wconv_kernel<<<(F_ * U_ + 255) / 256, 256, 0, stream>>>(W, wbf);
  sap_scores_kernel<<<(B_ * T_) / 128, 256, 0, stream>>>(x, wbf, bias, V, scores);
  sap_softmax_kernel<<<B_, 256, 0, stream>>>(scores, attn);
  sap_pool_kernel<<<B_, 128, 0, stream>>>(x, attn, out);
}
